// DynamicKernelSelection_26285199851631
// MI455X (gfx1250) — matrix operand order repaired and checked by reading, not compiled, not run on hardware
//
#include <hip/hip_runtime.h>
#include <hip/hip_bf16.h>

// ---------------- problem constants ----------------
constexpr int Bn = 16;
constexpr int Cn = 128;
constexpr int Hn = 128;
constexpr int Wn = 128;
constexpr int PLANE = Hn * Wn;                            // 16384
constexpr long long TENSOR = (long long)Bn * Cn * PLANE;  // 33,554,432

typedef __attribute__((ext_vector_type(16))) _Float16 v16h;
typedef __attribute__((ext_vector_type(8)))  float    v8f;

// ---------------- kernel 1: global average pool -> pooled[B*C] ----------------
__global__ void moe_pool(const float* __restrict__ x, float* __restrict__ pooled) {
  __shared__ float red[256];
  const int bc = blockIdx.x;                 // b*C + c
  const float4* plane = (const float4*)(x + (size_t)bc * PLANE);
  float s = 0.f;
  for (int i = threadIdx.x; i < PLANE / 4; i += 256) {
    float4 v = plane[i];
    s += v.x + v.y + v.z + v.w;
  }
  red[threadIdx.x] = s;
  __syncthreads();
  for (int off = 128; off > 0; off >>= 1) {
    if (threadIdx.x < off) red[threadIdx.x] += red[threadIdx.x + off];
    __syncthreads();
  }
  if (threadIdx.x == 0) pooled[bc] = red[0] * (1.0f / (float)PLANE);
}

// ---------------- kernel 2: gating matmul via WMMA + argmax routing ----------------
// logits[16x16] = pooled[16x128] x Wg[128x16]; Wg cols 0..1 = aw1, cols 2..4 = aw2.
// One wave32, EXEC all-ones; K=128 -> 4 chained v_wmma_f32_16x16x32_f16 (fully unrolled).
__global__ void moe_gate(const float* __restrict__ pooled,
                         const float* __restrict__ aw1, const float* __restrict__ ab1,
                         const float* __restrict__ aw2, const float* __restrict__ ab2,
                         int* __restrict__ idx1, int* __restrict__ idx2) {
  __shared__ float logits[16][17];
  const int lane = threadIdx.x;      // 0..31
  const int m    = lane & 15;        // A-row / B-col / D-col owned by this lane
  const int hi   = lane >> 4;        // half-wave select

  // Branch-free per-lane weight row for B (cols >= 5 read aw1 row 0 scaled by 0).
  const float* wrow = (m < 2) ? (aw1 + m * Cn)
                              : ((m < 5) ? (aw2 + (m - 2) * Cn) : aw1);
  const float wscale = (m < 5) ? 1.f : 0.f;

  v8f acc = {0.f, 0.f, 0.f, 0.f, 0.f, 0.f, 0.f, 0.f};

#pragma unroll
  for (int chunk = 0; chunk < 4; ++chunk) {
    const int base = chunk * 32;
    v16h a, bm;
    // A (16x32 f16): lane<16 holds K = base+{0..7,16..23}; lane>=16 holds K = base+{8..15,24..31}
#pragma unroll
    for (int j = 0; j < 8; ++j) {
      a[j]     = (_Float16)pooled[m * Cn + base + hi * 8 + j];
      a[j + 8] = (_Float16)pooled[m * Cn + base + 16 + hi * 8 + j];
    }
    // B (32x16 f16): column n = lane&15; lane<16 holds K rows base+{0..7,16..23}, lane>=16 base+{8..15,24..31}
#pragma unroll
    for (int j = 0; j < 16; ++j) {
      bm[j] = (_Float16)(wscale * wrow[base + hi * 8 + (j & 7) + (j >> 3) * 16]);
    }
    acc = __builtin_amdgcn_wmma_f32_16x16x32_f16(
        /*neg_a=*/false, a, /*neg_b=*/false, bm,
        /*c_mod=*/(short)0, acc, /*reuse_a=*/false, /*reuse_b=*/false);
  }

  // D layout: VGPR j -> row (j + 8*hi), col = lane&15
#pragma unroll
  for (int j = 0; j < 8; ++j) logits[j + 8 * hi][m] = acc[j];
  __syncthreads();

  if (lane < 16) {
    const int b = lane;
    // argmax(softmax(l)) == argmax(l); first-max tie-break like jnp.argmax
    float l0 = logits[b][0] + ab1[0];
    float l1 = logits[b][1] + ab1[1];
    idx1[b] = (l1 > l0) ? 1 : 0;
    float g0 = logits[b][2] + ab2[0];
    float g1 = logits[b][3] + ab2[1];
    float g2 = logits[b][4] + ab2[2];
    int bi = 0; float bv = g0;
    if (g1 > bv) { bv = g1; bi = 1; }
    if (g2 > bv) { bv = g2; bi = 2; }
    idx2[b] = bi;
  }
}

// ---------------- async global -> LDS helpers (gfx1250 native path) ----------------
__device__ __forceinline__ void async_row_b128(const float* gsrc, float* lds_dst) {
  // vDst = LDS byte address (low 32 bits of generic shared pointer),
  // vAddr = 64-bit global address; tracked by ASYNCcnt.
  unsigned lds_addr = (unsigned)(unsigned long long)(uintptr_t)lds_dst;
  unsigned long long gaddr = (unsigned long long)(uintptr_t)gsrc;
  asm volatile("global_load_async_to_lds_b128 %0, %1, off"
               :: "v"(lds_addr), "v"(gaddr) : "memory");
}
__device__ __forceinline__ void wait_async0() {
  asm volatile("s_wait_asynccnt 0" ::: "memory");
}

// ---------------- depthwise conv full-width strip (128 x 8 outputs / block) ----------------
// LDS row layout: [16 left-zero margin | 128 image cols | 16 right-zero margin] = 160 floats.
// In-bounds rows are DMA'd with 32-lane B128 async loads (one per row), halo zeros are
// ds_store'd to disjoint addresses, so no DS-vs-ASYNC ordering hazard.
template <int K, int DIL>
__device__ __forceinline__ void dwconv_strip(const float* __restrict__ in_plane,
                                             float* __restrict__ out_plane,
                                             const float* __restrict__ wk, float bias,
                                             int ty0, float* smem) {
  constexpr int HALO = (K / 2) * DIL;      // == reference padding
  constexpr int TH   = 8;                  // output rows per block
  constexpr int LH   = TH + 2 * HALO;      // tile rows
  constexpr int LWS  = 160;                // padded LDS row stride (floats)
  const int tid = threadIdx.x;
  float* tile = smem;
  float* wsm  = smem + LH * LWS;

  if (tid < K * K) wsm[tid] = wk[tid];

  // Zero left/right margins of every row (cols 0..15 and 144..159).
  for (int i = tid; i < LH * 32; i += 256) {
    const int lr = i >> 5, col = i & 31;
    tile[lr * LWS + (col < 16 ? col : 128 + col)] = 0.f;
  }
  // Zero interiors of vertically-OOB rows.
  for (int i = tid; i < LH * 128; i += 256) {
    const int lr = i >> 7;
    const int gy = ty0 - HALO + lr;
    if (gy < 0 || gy >= Hn) tile[lr * LWS + 16 + (i & 127)] = 0.f;
  }
  // Async DMA of in-bounds rows: 32 x float4 per row, 16B-aligned both sides.
  for (int i = tid; i < LH * 32; i += 256) {
    const int lr = i >> 5, g = i & 31;
    const int gy = ty0 - HALO + lr;
    if (gy >= 0 && gy < Hn) {
      async_row_b128(in_plane + gy * Wn + g * 4, tile + lr * LWS + 16 + g * 4);
    }
  }
  wait_async0();          // this wave's async transfers into LDS are complete
  __syncthreads();        // all waves' transfers visible

  const int lx  = tid & 127;   // output col
  const int ly0 = tid >> 7;    // 0..1; thread computes rows ly0, ly0+2, ly0+4, ly0+6
  const int cb  = 16 - HALO + lx;
  float acc0 = bias, acc1 = bias, acc2 = bias, acc3 = bias;

  for (int dy = 0; dy < K; ++dy) {
    const int rb = (ly0 + dy * DIL) * LWS + cb;
#pragma unroll
    for (int dx = 0; dx < K; ++dx) {
      const float wv = wsm[dy * K + dx];
      const int p = rb + dx * DIL;
      acc0 = fmaf(wv, tile[p],            acc0);
      acc1 = fmaf(wv, tile[p + 2 * LWS],  acc1);
      acc2 = fmaf(wv, tile[p + 4 * LWS],  acc2);
      acc3 = fmaf(wv, tile[p + 6 * LWS],  acc3);
    }
  }
  float* o = out_plane + (size_t)(ty0 + ly0) * Wn + lx;
  o[0]      = acc0;
  o[2 * Wn] = acc1;
  o[4 * Wn] = acc2;
  o[6 * Wn] = acc3;
}

// ---------------- kernel 3: stage-1 routed depthwise conv (k in {3,5}, dil=1) ----------------
__global__ void moe_conv1(const float* __restrict__ x, float* __restrict__ out1,
                          const float* __restrict__ w3, const float* __restrict__ b3,
                          const float* __restrict__ w5, const float* __restrict__ b5,
                          const int* __restrict__ idx1) {
  __shared__ __align__(16) float smem[12 * 160 + 32];   // k=5: LH=12 rows
  const int bc = blockIdx.z;
  const int b  = bc >> 7;                  // bc / C
  const int c  = bc & 127;                 // bc % C
  const int ty0 = blockIdx.y * 8;
  const float* in_plane  = x    + (size_t)bc * PLANE;
  float*       out_plane = out1 + (size_t)bc * PLANE;
  const int e = idx1[b];                   // uniform per block
  if (e == 0) dwconv_strip<3, 1>(in_plane, out_plane, w3 + c * 9,  b3[c], ty0, smem);
  else        dwconv_strip<5, 1>(in_plane, out_plane, w5 + c * 25, b5[c], ty0, smem);
}

// ---------------- kernel 4: stage-2 routed depthwise conv (k in {7,9,11}, dil=3) ----------------
__global__ void moe_conv2(const float* __restrict__ out1, float* __restrict__ out2,
                          const float* __restrict__ w7,  const float* __restrict__ b7,
                          const float* __restrict__ w9,  const float* __restrict__ b9,
                          const float* __restrict__ w11, const float* __restrict__ b11,
                          const int* __restrict__ idx2) {
  __shared__ __align__(16) float smem[38 * 160 + 128];  // k=11,dil=3: LH=38 rows (~24.3 KB)
  const int bc = blockIdx.z;
  const int b  = bc >> 7;
  const int c  = bc & 127;
  const int ty0 = blockIdx.y * 8;
  const float* in_plane  = out1 + (size_t)bc * PLANE;
  float*       out_plane = out2 + (size_t)bc * PLANE;
  const int e = idx2[b];                   // uniform per block
  if (e == 0)      dwconv_strip<7,  3>(in_plane, out_plane, w7  + c * 49,  b7[c],  ty0, smem);
  else if (e == 1) dwconv_strip<9,  3>(in_plane, out_plane, w9  + c * 81,  b9[c],  ty0, smem);
  else             dwconv_strip<11, 3>(in_plane, out_plane, w11 + c * 121, b11[c], ty0, smem);
}

// ---------------- host launcher ----------------
extern "C" void kernel_launch(void* const* d_in, const int* in_sizes, int n_in,
                              void* d_out, int out_size, void* d_ws, size_t ws_size,
                              hipStream_t stream) {
  const float* x    = (const float*)d_in[0];
  const float* aw1  = (const float*)d_in[1];
  const float* ab1  = (const float*)d_in[2];
  const float* aw2  = (const float*)d_in[3];
  const float* ab2  = (const float*)d_in[4];
  const float* w1_3 = (const float*)d_in[5];
  const float* b1_3 = (const float*)d_in[6];
  const float* w1_5 = (const float*)d_in[7];
  const float* b1_5 = (const float*)d_in[8];
  const float* w2_7 = (const float*)d_in[9];
  const float* b2_7 = (const float*)d_in[10];
  const float* w2_9 = (const float*)d_in[11];
  const float* b2_9 = (const float*)d_in[12];
  const float* w2_11= (const float*)d_in[13];
  const float* b2_11= (const float*)d_in[14];

  float* out1 = (float*)d_out;
  float* out2 = out1 + TENSOR;

  float* pooled = (float*)d_ws;                              // B*C floats
  int*   idx1   = (int*)((char*)d_ws + Bn * Cn * sizeof(float));
  int*   idx2   = idx1 + Bn;

  // 1) pooled = mean(x, H, W)
  moe_pool<<<Bn * Cn, 256, 0, stream>>>(x, pooled);

  // 2) WMMA gating + argmax routing (one wave32)
  moe_gate<<<1, 32, 0, stream>>>(pooled, aw1, ab1, aw2, ab2, idx1, idx2);

  // 3) stage-1 routed depthwise conv -> out1
  dim3 grid(1, Hn / 8, Bn * Cn);
  moe_conv1<<<grid, 256, 0, stream>>>(x, out1, w1_3, b1_3, w1_5, b1_5, idx1);

  // 4) stage-2 routed dilated depthwise conv on out1 -> out2
  moe_conv2<<<grid, 256, 0, stream>>>(out1, out2, w2_7, b2_7, w2_9, b2_9, w2_11, b2_11, idx2);
}
